// AtomEdgeModel_11106785427534
// MI455X (gfx1250) — matrix operand order repaired and checked by reading, not compiled, not run on hardware
//
#include <hip/hip_runtime.h>
#include <hip/hip_bf16.h>

// ---------------------------------------------------------------------------
// GATv2 x4 + edge MLP for gfx1250 (MI455X).  wave32; WMMA f32_16x16x32_f16.
// ---------------------------------------------------------------------------

#define LREL 0.01f   // inter-layer LeakyReLU
#define GATS 0.2f    // GATv2 internal LeakyReLU

typedef __attribute__((ext_vector_type(16))) _Float16 v16h;
typedef __attribute__((ext_vector_type(8)))  _Float16 v8h;
typedef __attribute__((ext_vector_type(8)))  float    v8f;

union V16U { v16h v; v8h h[2]; };

static __device__ __forceinline__ unsigned encF(float f) {
  unsigned u = __float_as_uint(f);
  return (u & 0x80000000u) ? ~u : (u | 0x80000000u);   // order-preserving map
}
static __device__ __forceinline__ float decF(unsigned u) {
  return __uint_as_float((u & 0x80000000u) ? (u & 0x7fffffffu) : ~u);
}

// ----------------------------- utility kernels -----------------------------

__global__ void fill_f32_k(float* p, float v, long n) {
  long t = (long)blockIdx.x * blockDim.x + threadIdx.x;
  if (t < n) p[t] = v;
}
__global__ void fill_u32_k(unsigned* p, unsigned v, long n) {
  long t = (long)blockIdx.x * blockDim.x + threadIdx.x;
  if (t < n) p[t] = v;
}
__global__ void sum_atomic_k(const float* __restrict__ p, float* acc, int n) {
  int t = blockIdx.x * blockDim.x + threadIdx.x;
  float v = (t < n) ? p[t] : 0.f;
#pragma unroll
  for (int o = 16; o > 0; o >>= 1) v += __shfl_down(v, o, 32);
  if ((threadIdx.x & 31) == 0) atomicAdd(acc, v);
}
// f32 [M,K] -> f16 [M,Kp] zero-padded (Kp multiple of 32 => aligned WMMA loads)
__global__ void stage_a_f16_k(const float* __restrict__ s, _Float16* __restrict__ d,
                              long M, int K, int Kp) {
  long t = (long)blockIdx.x * blockDim.x + threadIdx.x;
  if (t >= M * (long)Kp) return;
  long r = t / Kp; int c = (int)(t - r * Kp);
  d[t] = (c < K) ? (_Float16)s[r * K + c] : (_Float16)0.f;
}
// weight f32 [Nout,K] -> f16 [Np,Kp] zero-padded rows+cols
__global__ void stage_w_f16_k(const float* __restrict__ s, _Float16* __restrict__ d,
                              int Nout, int Np, int K, int Kp) {
  int t = blockIdx.x * blockDim.x + threadIdx.x;
  if (t >= Np * Kp) return;
  int r = t / Kp, c = t - r * Kp;
  d[t] = (r < Nout && c < K) ? (_Float16)s[r * K + c] : (_Float16)0.f;
}
// raw x -> xc_h columns [0,32)
__global__ void stage_x_to_xc_k(const float* __restrict__ x, _Float16* __restrict__ xc, long n) {
  long t = (long)blockIdx.x * blockDim.x + threadIdx.x;
  if (t >= n) return;
  long r = t >> 5; int c = (int)(t & 31);
  xc[r * 282 + c] = (_Float16)x[t];
}
__global__ void init_out_k(float* __restrict__ out, const float* __restrict__ bias,
                           long total, int HC) {
  long t = (long)blockIdx.x * blockDim.x + threadIdx.x;
  if (t < total) out[t] = bias[t % HC];
}
// lrelu + optional f32 writeback + optional f16 write into xc_h
__global__ void finish_layer_k(const float* __restrict__ acc, float* __restrict__ dstF,
                               _Float16* __restrict__ xc, int xcOff, long total, int HC) {
  long t = (long)blockIdx.x * blockDim.x + threadIdx.x;
  if (t >= total) return;
  long r = t / HC; int c = (int)(t - r * HC);
  float v = acc[t];
  v = v > 0.f ? v : LREL * v;
  if (dstF) dstF[t] = v;
  if (xc) xc[r * 282 + xcOff + c] = (_Float16)v;
}

// ------------------------------- WMMA GEMM ---------------------------------
// OUT[m, n] = sum_k A[m,k] * W[n,k] + bias[n]     (A: f16 [M,Kp], W: f16 [Np,Kp])
// One wave per 16x16 output tile.  M % 16 == 0 (N = 50000 = 3125*16).
__global__ void __launch_bounds__(32)
wmma_gemm_nt_k(const _Float16* __restrict__ A, const _Float16* __restrict__ W,
               const float* __restrict__ bias, float* __restrict__ out,
               int Nout, int Kp, int outStride) {
  int lane = threadIdx.x & 31;
  int g = lane >> 4, m = lane & 15;
  long m0 = (long)blockIdx.x * 16;
  int  n0 = blockIdx.y * 16;
  const _Float16* arow = A + (m0 + m) * (long)Kp;
  const _Float16* brow = W + (long)(n0 + m) * Kp;
  v8f c = {0.f, 0.f, 0.f, 0.f, 0.f, 0.f, 0.f, 0.f};
  for (int kk = 0; kk < Kp; kk += 32) {
    V16U a, b;
    a.v    = *(const v16h*)(arow + kk + g * 16);      // K = kk+g*16  .. +15
    b.v    = *(const v16h*)(brow + kk + g * 16);      // K = kk+g*16  .. +15
    c = __builtin_amdgcn_wmma_f32_16x16x32_f16(false, a.v, false, b.v,
                                               (short)0, c, false, false);
  }
  int col = n0 + m;
  if (col < Nout) {
    float bv = bias ? bias[col] : 0.f;
#pragma unroll
    for (int j = 0; j < 8; ++j)
      out[(m0 + g * 8 + j) * (long)outStride + col] = c[j] + bv;
  }
}

// ------------------------------ GATv2 kernels ------------------------------
// score[e,h] = sum_c lrelu(xl[src]+xr[dst]+attr*We, 0.2)[h,c] * att[h,c]
__global__ void gat_score_k(const float* __restrict__ xl, const float* __restrict__ xr,
                            const int* __restrict__ src, const int* __restrict__ dst,
                            const float* __restrict__ eattr, const float* __restrict__ asum,
                            float invE0, const float* __restrict__ We,
                            const float* __restrict__ att, float* __restrict__ score,
                            unsigned* __restrict__ nmax, int E0, int Etot, int H, int C) {
  int e = blockIdx.x * blockDim.x + threadIdx.x;
  if (e >= Etot) return;
  int s, d; float a;
  if (e < E0) { s = src[e]; d = dst[e]; a = eattr[e]; }
  else        { s = d = e - E0; a = asum[0] * invE0; }   // self loop, mean attr
  int HC = H * C;
  const float* xls = xl + (long)s * HC;
  const float* xrd = xr + (long)d * HC;
  for (int h = 0; h < H; ++h) {
    float sc = 0.f;
    for (int cc = 0; cc < C; ++cc) {
      int i = h * C + cc;
      float z = xls[i] + xrd[i] + a * We[i];
      z = z > 0.f ? z : GATS * z;
      sc += z * att[i];
    }
    score[(long)e * H + h] = sc;
    atomicMax(&nmax[d * H + h], encF(sc));
  }
}
__global__ void gat_exp_k(const int* __restrict__ dst, float* __restrict__ score,
                          const unsigned* __restrict__ nmax, float* __restrict__ nsum,
                          int E0, int Etot, int H) {
  int e = blockIdx.x * blockDim.x + threadIdx.x;
  if (e >= Etot) return;
  int d = (e < E0) ? dst[e] : (e - E0);
  for (int h = 0; h < H; ++h) {
    float mx = decF(nmax[d * H + h]);
    float v = __expf(score[(long)e * H + h] - mx);
    score[(long)e * H + h] = v;
    atomicAdd(&nsum[d * H + h], v);
  }
}
__global__ void gat_agg_k(const float* __restrict__ xl, const int* __restrict__ src,
                          const int* __restrict__ dst, const float* __restrict__ score,
                          const float* __restrict__ nsum, float* __restrict__ out,
                          int E0, int Etot, int H, int C) {
  int e = blockIdx.x * blockDim.x + threadIdx.x;
  if (e >= Etot) return;
  int s, d;
  if (e < E0) { s = src[e]; d = dst[e]; } else { s = d = e - E0; }
  int HC = H * C;
  for (int h = 0; h < H; ++h) {
    float alpha = score[(long)e * H + h] / nsum[d * H + h];
    const float* xls = xl + (long)s * HC + h * C;
    float* od = out + (long)d * HC + h * C;
    for (int cc = 0; cc < C; ++cc) atomicAdd(&od[cc], alpha * xls[cc]);
  }
}

// ----------------------- fused edge MLP (e1 -> e2 -> e3) -------------------
// Per block (1 wave): 16 edges.  Gather ee[16][576] f16 into LDS, WMMA against
// W1h [112,576], lrelu, then per-edge 100->10->1 tail, atomicAdd block sum.
__global__ void __launch_bounds__(32)
edge_mlp_k(const _Float16* __restrict__ xc, const int* __restrict__ src,
           const int* __restrict__ dst, const float* __restrict__ attr,
           const _Float16* __restrict__ W1h, const float* __restrict__ b1,
           const float* __restrict__ W2, const float* __restrict__ b2,
           const float* __restrict__ W3, const float* __restrict__ b3,
           float* __restrict__ out_sum) {
  __shared__ __align__(16) _Float16 ee[16][576];
  __shared__ float h1[16][112];
  int lane = threadIdx.x & 31;
  int g = lane >> 4, m = lane & 15;
  long e0 = (long)blockIdx.x * 16;
  // gather 16 edge feature rows: [xc[s](282) | xc[d](282) | attr | 0-pad]
  for (int r = 0; r < 16; ++r) {
    int s = src[e0 + r], d = dst[e0 + r];
    float av = attr[e0 + r];
    for (int cc = lane; cc < 576; cc += 32) {
      _Float16 v;
      if (cc < 282)       v = xc[(long)s * 282 + cc];
      else if (cc < 564)  v = xc[(long)d * 282 + (cc - 282)];
      else if (cc == 564) v = (_Float16)av;
      else                v = (_Float16)0.f;
      ee[r][cc] = v;
    }
  }
  __syncthreads();
  // e1: 16x576 @ 576x112 via WMMA (A from LDS, B from global)
  for (int nt = 0; nt < 7; ++nt) {
    const _Float16* brow = W1h + (long)(nt * 16 + m) * 576;
    v8f c = {0.f, 0.f, 0.f, 0.f, 0.f, 0.f, 0.f, 0.f};
    for (int kk = 0; kk < 576; kk += 32) {
      V16U a, b;
      a.v    = *(const v16h*)&ee[m][kk + g * 16];
      b.v    = *(const v16h*)(brow + kk + g * 16);
      c = __builtin_amdgcn_wmma_f32_16x16x32_f16(false, a.v, false, b.v,
                                                 (short)0, c, false, false);
    }
    int col = nt * 16 + m;
    float bv = (col < 100) ? b1[col] : 0.f;
#pragma unroll
    for (int j = 0; j < 8; ++j) {
      float v = c[j] + bv;
      h1[g * 8 + j][col] = v > 0.f ? v : LREL * v;
    }
  }
  __syncthreads();
  // e2 (100->10, lrelu) + e3 (10->1), one edge per lane
  if (lane < 16) {
    float h2[10];
    for (int o = 0; o < 10; ++o) {
      float acc = b2[o];
      for (int k = 0; k < 100; ++k) acc += W2[o * 100 + k] * h1[lane][k];
      h2[o] = acc > 0.f ? acc : LREL * acc;
    }
    float y = b3[0];
#pragma unroll
    for (int o = 0; o < 10; ++o) y += W3[o] * h2[o];
    atomicAdd(out_sum, y);
  }
}

// --------------------------------- launcher --------------------------------

extern "C" void kernel_launch(void* const* d_in, const int* in_sizes, int n_in,
                              void* d_out, int out_size, void* d_ws, size_t ws_size,
                              hipStream_t stream) {
  (void)n_in; (void)out_size; (void)ws_size;
  // inputs in setup_inputs() order, params flattened depth-first
  const float* x       = (const float*)d_in[0];
  const int*   er_idx  = (const int*)  d_in[1];
  const float* er_attr = (const float*)d_in[2];
  const int*   ep_idx  = (const int*)  d_in[3];
  const float* ep_attr = (const float*)d_in[4];
  const int*   ei_idx  = (const int*)  d_in[5];
  const float* ei_attr = (const float*)d_in[6];
  auto F = [&](int i) { return (const float*)d_in[i]; };

  const int N    = in_sizes[0] / 32;      // 50000
  const int E_er = in_sizes[1] / 2;       // 400000
  const int E_ep = in_sizes[3] / 2;       // 800000
  const int E_ei = in_sizes[5] / 2;       // 200000

  // ---- workspace bump allocator ----
  size_t off = 0;
  auto alloc = [&](size_t bytes) {
    void* p = (char*)d_ws + off;
    off = (off + bytes + 255) & ~(size_t)255;
    return p;
  };
  _Float16* xcH  = (_Float16*)alloc((size_t)N * 282 * 2);
  _Float16* Ah   = (_Float16*)alloc((size_t)N * 160 * 2);
  _Float16* WlH  = (_Float16*)alloc(160 * 160 * 2);
  _Float16* WrH  = (_Float16*)alloc(160 * 160 * 2);
  _Float16* W1H  = (_Float16*)alloc(112 * 576 * 2);
  float* xl   = (float*)alloc((size_t)N * 150 * 4);
  float* xr   = (float*)alloc((size_t)N * 150 * 4);
  float* accB = (float*)alloc((size_t)N * 150 * 4);
  float* pep  = (float*)alloc((size_t)N * 150 * 4);
  float* h1b  = (float*)alloc((size_t)N * 150 * 4);
  float* h2b  = (float*)alloc((size_t)N * 100 * 4);
  float* scoreB = (float*)alloc(((size_t)E_ep + N) * 3 * 4);
  unsigned* nmax = (unsigned*)alloc((size_t)N * 3 * 4);
  float*    nsum = (float*)   alloc((size_t)N * 3 * 4);
  float*    asums = (float*)  alloc(2 * 4);

  const int TB = 256;
  auto gr = [&](long n) { return dim3((unsigned)((n + TB - 1) / TB)); };

  // edge-attr sums (for self-loop fill_value='mean')
  fill_f32_k<<<dim3(1), 32, 0, stream>>>(asums, 0.f, 2);
  sum_atomic_k<<<gr(E_er), TB, 0, stream>>>(er_attr, asums + 0, E_er);
  sum_atomic_k<<<gr(E_ep), TB, 0, stream>>>(ep_attr, asums + 1, E_ep);

  // raw x into xc_h cols [0,32)
  stage_x_to_xc_k<<<gr((long)N * 32), TB, 0, stream>>>(x, xcH, (long)N * 32);

  auto run_gat = [&](const float* Xf, int Kin,
                     const float* Wl, const float* bl, const float* Wr, const float* br,
                     const float* We, const float* att, const float* bias,
                     int H, int C,
                     const int* esrc, const int* edst, const float* eattr,
                     const float* asum, int E0, float* dstF, int xcOff) {
    int HC = H * C;
    int Kp = ((Kin + 31) / 32) * 32;
    int Np = ((HC + 15) / 16) * 16;
    long Etot = (long)E0 + N;
    stage_a_f16_k<<<gr((long)N * Kp), TB, 0, stream>>>(Xf, Ah, N, Kin, Kp);
    stage_w_f16_k<<<gr((long)Np * Kp), TB, 0, stream>>>(Wl, WlH, HC, Np, Kin, Kp);
    stage_w_f16_k<<<gr((long)Np * Kp), TB, 0, stream>>>(Wr, WrH, HC, Np, Kin, Kp);
    wmma_gemm_nt_k<<<dim3(N / 16, Np / 16), 32, 0, stream>>>(Ah, WlH, bl, xl, HC, Kp, HC);
    wmma_gemm_nt_k<<<dim3(N / 16, Np / 16), 32, 0, stream>>>(Ah, WrH, br, xr, HC, Kp, HC);
    fill_u32_k<<<gr((long)N * H), TB, 0, stream>>>(nmax, 0u, (long)N * H);
    fill_f32_k<<<gr((long)N * H), TB, 0, stream>>>(nsum, 0.f, (long)N * H);
    gat_score_k<<<gr(Etot), TB, 0, stream>>>(xl, xr, esrc, edst, eattr, asum,
                                             1.0f / (float)E0, We, att, scoreB, nmax,
                                             E0, (int)Etot, H, C);
    gat_exp_k<<<gr(Etot), TB, 0, stream>>>(edst, scoreB, nmax, nsum, E0, (int)Etot, H);
    init_out_k<<<gr((long)N * HC), TB, 0, stream>>>(accB, bias, (long)N * HC, HC);
    gat_agg_k<<<gr(Etot), TB, 0, stream>>>(xl, esrc, edst, scoreB, nsum, accB,
                                           E0, (int)Etot, H, C);
    finish_layer_k<<<gr((long)N * HC), TB, 0, stream>>>(
        accB, dstF, xcOff >= 0 ? xcH : (_Float16*)nullptr,
        xcOff >= 0 ? xcOff : 0, (long)N * HC, HC);
  };

  const int* er_src = er_idx;            const int* er_dst = er_idx + E_er;
  const int* ep_src = ep_idx;            const int* ep_dst = ep_idx + E_ep;
  const int* ei_src = ei_idx;            const int* ei_dst = ei_idx + E_ei;

  // g1: 32 -> 3x50 = 150   (params d_in[7..13])
  run_gat(x,   32, F(7),  F(8),  F(9),  F(10), F(11), F(12), F(13), 3, 50,
          er_src, er_dst, er_attr, asums + 0, E_er, pep, /*xc col*/32);
  // g2: 150 -> 2x75 = 150  (d_in[14..20])
  run_gat(pep, 150, F(14), F(15), F(16), F(17), F(18), F(19), F(20), 2, 75,
          ep_src, ep_dst, ep_attr, asums + 1, E_ep, h1b, -1);
  // g3: 150 -> 2x50 = 100  (d_in[21..27])
  run_gat(h1b, 150, F(21), F(22), F(23), F(24), F(25), F(26), F(27), 2, 50,
          ep_src, ep_dst, ep_attr, asums + 1, E_ep, h2b, -1);
  // g4: 100 -> 1x100 = 100 (d_in[28..34]); output only feeds xc cols [182,282)
  run_gat(h2b, 100, F(28), F(29), F(30), F(31), F(32), F(33), F(34), 1, 100,
          ep_src, ep_dst, ep_attr, asums + 1, E_ep, (float*)nullptr, 182);

  // edge MLP: e1 W [100,565] -> f16 [112,576]
  stage_w_f16_k<<<gr(112 * 576), TB, 0, stream>>>(F(35), W1H, 100, 112, 565, 576);
  fill_f32_k<<<dim3(1), 32, 0, stream>>>((float*)d_out, 0.f, 1);
  edge_mlp_k<<<dim3(E_ei / 16), 32, 0, stream>>>(xcH, ei_src, ei_dst, ei_attr,
                                                 W1H, F(36), F(37), F(38),
                                                 F(39), F(40), (float*)d_out);
}
